// ImprovedLookupFFN_68719476736164
// MI455X (gfx1250) — matrix operand order repaired and checked by reading, not compiled, not run on hardware
//
#include <hip/hip_runtime.h>
#include <hip/hip_fp16.h>

typedef __attribute__((ext_vector_type(16))) _Float16 v16h;
typedef __attribute__((ext_vector_type(8)))  _Float16 v8h;
typedef __attribute__((ext_vector_type(4)))  _Float16 v4h;
typedef __attribute__((ext_vector_type(8)))  float    v8f;
typedef __attribute__((ext_vector_type(4)))  float    v4f;

#define NROWS 262144
#define DIN   256
#define NBKT  512
#define NCLS  256

// Per-wave LDS layout (bytes)
#define PROJ_BYTES (16*512*4)   // f32 proj tile
#define WTS_BYTES  (16*512*2)   // f16 softmax weights
#define MID_BYTES  (16*256*2)   // f16 GEMM2 output
#define RS_BYTES   64           // f32 1/rowsum[16]
#define WAVE_LDS   (PROJ_BYTES + WTS_BYTES + MID_BYTES + RS_BYTES)  // 57408
#define WPB        4            // waves per block
#define SMEM_BYTES (WAVE_LDS * WPB)                                  // 229632 <= 320KB

// Packed f16 weight sizes (halves)
#define RP_HALVES (NBKT * DIN)   // 131072
#define LP_HALVES (NBKT * NCLS)  // 131072
#define WP_HALVES (NCLS * NCLS)  // 65536

// ---------------------------------------------------------------------------
// Pack kernel: convert fp32 weights to f16 in WMMA B-fragment order.
// B fragment (K=32 x N=16) per-lane layout assumed:
//   lane l: n = nt*16 + (l&15), holds K = kt*32 + (l>>4)*8 + (j&7) + (j>>3)*16, j=0..15
// so each lane's 16 halves are one contiguous 32B v16h load.
// ---------------------------------------------------------------------------
__global__ void pack_weights(const float* __restrict__ R,
                             const float* __restrict__ L,
                             const float* __restrict__ W,
                             _Float16* __restrict__ Rp,
                             _Float16* __restrict__ Lp,
                             _Float16* __restrict__ Wp) {
  int i = blockIdx.x * blockDim.x + threadIdx.x;
  if (i < RP_HALVES) {
    // GEMM1 B = R^T scaled by 1/TEMPERATURE (=2.0): B[k=d][n=bucket] = 2*R[n][d]
    int j = i & 15, lane = (i >> 4) & 31, fi = i >> 9;
    int kt = fi & 7, nt = fi >> 3;
    int n = nt * 16 + (lane & 15);
    int k = kt * 32 + (lane >> 4) * 8 + (j & 7) + (j >> 3) * 16;
    Rp[i] = (_Float16)(R[n * DIN + k] * 2.0f);
  } else if (i < RP_HALVES + LP_HALVES) {
    // GEMM2 B = L: B[k=bucket][n=class] = L[k][n]
    int ii = i - RP_HALVES;
    int j = ii & 15, lane = (ii >> 4) & 31, fi = ii >> 9;
    int kt = fi & 15, nt = fi >> 4;
    int n = nt * 16 + (lane & 15);
    int k = kt * 32 + (lane >> 4) * 8 + (j & 7) + (j >> 3) * 16;
    Lp[ii] = (_Float16)L[k * NCLS + n];
  } else if (i < RP_HALVES + LP_HALVES + WP_HALVES) {
    // GEMM3 B = post_w^T: B[k=c][n=o] = post_w[o][c]
    int ii = i - RP_HALVES - LP_HALVES;
    int j = ii & 15, lane = (ii >> 4) & 31, fi = ii >> 9;
    int kt = fi & 7, nt = fi >> 3;
    int n = nt * 16 + (lane & 15);
    int k = kt * 32 + (lane >> 4) * 8 + (j & 7) + (j >> 3) * 16;
    Wp[ii] = (_Float16)W[n * NCLS + k];
  }
}

// Build a 16x32 A-fragment from an LDS f16 row-major tile.
// ISA A layout: lane holds M=lane%16; K runs [base, base+8) and [base+16, base+24)
// with base = kt*32 + (lane>>4)*8. Both runs are contiguous 16B-aligned -> ds_load_b128.
__device__ __forceinline__ v16h lds_afrag(const _Float16* __restrict__ base) {
  v8h lo = *(const v8h*)(base);        // ds_load_b128
  v8h hi = *(const v8h*)(base + 16);   // ds_load_b128
  return __builtin_shufflevector(lo, hi, 0, 1, 2, 3, 4, 5, 6, 7,
                                 8, 9, 10, 11, 12, 13, 14, 15);
}

// ---------------------------------------------------------------------------
// Fused: proj = x@R^T/T -> softmax -> @L -> @post_w^T + b
// One wave per 16-row tile; 4 waves (64 rows) per block.
// ---------------------------------------------------------------------------
__global__ __launch_bounds__(128, 1)
void fused_lookup_ffn(const float* __restrict__ x,
                      const _Float16* __restrict__ Rp,
                      const _Float16* __restrict__ Lp,
                      const _Float16* __restrict__ Wp,
                      const float* __restrict__ bias,
                      float* __restrict__ out) {
  extern __shared__ char smem[];
  const int lane = threadIdx.x & 31;
  const int wave = threadIdx.x >> 5;
  char* wb = smem + wave * WAVE_LDS;
  float*    proj = (float*)wb;
  _Float16* wts  = (_Float16*)(wb + PROJ_BYTES);
  _Float16* mid  = (_Float16*)(wb + PROJ_BYTES + WTS_BYTES);
  float*    rsum = (float*)(wb + PROJ_BYTES + WTS_BYTES + MID_BYTES);

  const int rowBase = blockIdx.x * (16 * WPB) + wave * 16;
  const int m  = lane & 15;   // A-matrix M / C-matrix N lane mapping
  const int kh = lane >> 4;   // which K-half this lane holds

  // ---- Build all 8 A-fragments (16x32 f16) for GEMM1 from x, in registers.
  // x is streamed once: non-temporal loads keep L2 reserved for packed weights.
  v16h afrag[8];
  const float* xrow = x + (size_t)(rowBase + m) * DIN;
  #pragma unroll
  for (int kt = 0; kt < 8; ++kt) {
    const v4f* p0 = (const v4f*)(xrow + kt * 32 + kh * 8);
    const v4f* p1 = (const v4f*)(xrow + kt * 32 + 16 + kh * 8);
    v4f a0 = __builtin_nontemporal_load(p0);
    v4f a1 = __builtin_nontemporal_load(p0 + 1);
    v4f b0 = __builtin_nontemporal_load(p1);
    v4f b1 = __builtin_nontemporal_load(p1 + 1);
    v16h f;
    f[0]  = (_Float16)a0[0]; f[1]  = (_Float16)a0[1]; f[2]  = (_Float16)a0[2]; f[3]  = (_Float16)a0[3];
    f[4]  = (_Float16)a1[0]; f[5]  = (_Float16)a1[1]; f[6]  = (_Float16)a1[2]; f[7]  = (_Float16)a1[3];
    f[8]  = (_Float16)b0[0]; f[9]  = (_Float16)b0[1]; f[10] = (_Float16)b0[2]; f[11] = (_Float16)b0[3];
    f[12] = (_Float16)b1[0]; f[13] = (_Float16)b1[1]; f[14] = (_Float16)b1[2]; f[15] = (_Float16)b1[3];
    afrag[kt] = f;
  }

  // ---- GEMM1: proj[16x512] = x_tile @ (R^T / T)
  for (int nt = 0; nt < 32; ++nt) {
    v8f acc = {};
    #pragma unroll
    for (int kt = 0; kt < 8; ++kt) {
      v16h b = *(const v16h*)(Rp + ((size_t)(nt * 8 + kt) * 32 + lane) * 16);
      acc = __builtin_amdgcn_wmma_f32_16x16x32_f16(false, afrag[kt], false, b,
                                                   (short)0, acc, false, false);
    }
    // C layout: lane l -> N = nt*16 + (l&15); VGPR v -> M = v + 8*(l>>4)
    #pragma unroll
    for (int v = 0; v < 8; ++v)
      proj[(v + kh * 8) * 512 + nt * 16 + m] = acc[v];
  }
  __syncthreads();

  // ---- Softmax over 512 buckets per row (all 32 lanes active: half-row each)
  {
    const int row = lane & 15;
    const int half = lane >> 4;
    const v4f* pr4 = (const v4f*)(proj + row * 512 + half * 256);
    float mx = -3.4e38f;
    #pragma unroll 4
    for (int i = 0; i < 64; ++i) {
      v4f f = pr4[i];
      mx = fmaxf(mx, fmaxf(fmaxf(f[0], f[1]), fmaxf(f[2], f[3])));
    }
    mx = fmaxf(mx, __shfl_xor(mx, 16, 32));
    float s = 0.0f;
    v4h* wr4 = (v4h*)(wts + row * 512 + half * 256);
    #pragma unroll 4
    for (int i = 0; i < 64; ++i) {
      v4f f = pr4[i];
      float e0 = __expf(f[0] - mx), e1 = __expf(f[1] - mx);
      float e2 = __expf(f[2] - mx), e3 = __expf(f[3] - mx);
      s += (e0 + e1) + (e2 + e3);
      v4h h; h[0] = (_Float16)e0; h[1] = (_Float16)e1; h[2] = (_Float16)e2; h[3] = (_Float16)e3;
      wr4[i] = h;
    }
    s += __shfl_xor(s, 16, 32);
    if (half == 0) rsum[row] = 1.0f / s;   // apply after GEMM2 (linear)
  }
  __syncthreads();

  // ---- GEMM2: mid[16x256] = softmax(proj) @ L  (scale by 1/rowsum on epilogue)
  float rs[8];
  #pragma unroll
  for (int v = 0; v < 8; ++v) rs[v] = rsum[v + kh * 8];
  for (int nt = 0; nt < 16; ++nt) {
    v8f acc = {};
    #pragma unroll 4
    for (int kt = 0; kt < 16; ++kt) {
      v16h a = lds_afrag(wts + m * 512 + kt * 32 + kh * 8);
      v16h b = *(const v16h*)(Lp + ((size_t)(nt * 16 + kt) * 32 + lane) * 16);
      acc = __builtin_amdgcn_wmma_f32_16x16x32_f16(false, a, false, b,
                                                   (short)0, acc, false, false);
    }
    #pragma unroll
    for (int v = 0; v < 8; ++v)
      mid[(v + kh * 8) * 256 + nt * 16 + m] = (_Float16)(acc[v] * rs[v]);
  }
  __syncthreads();

  // ---- GEMM3: out[16x256] = mid @ post_w^T + bias (non-temporal streaming store)
  for (int nt = 0; nt < 16; ++nt) {
    v8f acc = {};
    #pragma unroll
    for (int kt = 0; kt < 8; ++kt) {
      v16h a = lds_afrag(mid + m * 256 + kt * 32 + kh * 8);
      v16h b = *(const v16h*)(Wp + ((size_t)(nt * 8 + kt) * 32 + lane) * 16);
      acc = __builtin_amdgcn_wmma_f32_16x16x32_f16(false, a, false, b,
                                                   (short)0, acc, false, false);
    }
    float bn = bias[nt * 16 + m];
    float* orow = out + (size_t)rowBase * NCLS + nt * 16 + m;
    #pragma unroll
    for (int v = 0; v < 8; ++v)
      __builtin_nontemporal_store(acc[v] + bn, orow + (size_t)(v + kh * 8) * NCLS);
  }
}

extern "C" void kernel_launch(void* const* d_in, const int* in_sizes, int n_in,
                              void* d_out, int out_size, void* d_ws, size_t ws_size,
                              hipStream_t stream) {
  (void)in_sizes; (void)n_in; (void)out_size; (void)ws_size;
  const float* x      = (const float*)d_in[0];
  const float* R      = (const float*)d_in[1];
  const float* L      = (const float*)d_in[2];
  const float* post_w = (const float*)d_in[3];
  const float* post_b = (const float*)d_in[4];
  float* out = (float*)d_out;

  _Float16* Rp = (_Float16*)d_ws;            // 256 KB
  _Float16* Lp = Rp + RP_HALVES;             // 256 KB
  _Float16* Wp = Lp + LP_HALVES;             // 128 KB

  // 1) Pack fp32 weights into f16 WMMA B-fragment layout (tiny, L2-resident).
  const int packElems = RP_HALVES + LP_HALVES + WP_HALVES;  // 327680
  pack_weights<<<(packElems + 255) / 256, 256, 0, stream>>>(R, L, post_w, Rp, Lp, Wp);

  // 2) Fused GEMM -> softmax -> GEMM -> GEMM, 64 rows per block.
  const int blocks = NROWS / (16 * WPB);  // 4096
  fused_lookup_ffn<<<blocks, 32 * WPB, SMEM_BYTES, stream>>>(x, Rp, Lp, Wp, post_b, out);
}
